// Attention_77610059039168
// MI455X (gfx1250) — matrix operand order repaired and checked by reading, not compiled, not run on hardware
//
#include <hip/hip_runtime.h>
#include <hip/hip_bf16.h>

typedef __attribute__((ext_vector_type(16))) __bf16 v16bf;
typedef __attribute__((ext_vector_type(8)))  float  v8f;

union FragAB { v16bf v; uint4 q[2]; };
union FragC  { v8f   v; float f[8]; };

#define NUM_HEADS 12
#define SEQ   197
#define NP    224            // SEQ padded to 14 tiles of 16
#define CDIM  768
#define HD    64
#define BATCH 64
#define MROWS (BATCH * SEQ)  // 12608 (multiple of 16)

// ---------------- f32 -> bf16 conversion ----------------
struct bf16x4 { __bf16 a, b, c, d; };

__global__ void cvt_f32_bf16(const float* __restrict__ in,
                             __bf16* __restrict__ out, int n4) {
  int i = blockIdx.x * blockDim.x + threadIdx.x;
  if (i < n4) {
    float4 f = ((const float4*)in)[i];
    bf16x4 r;
    r.a = (__bf16)f.x; r.b = (__bf16)f.y; r.c = (__bf16)f.z; r.d = (__bf16)f.w;
    ((bf16x4*)out)[i] = r;
  }
}

// ---- async global->LDS copy of 16B per lane (CDNA5 TDM-lite path) ----
__device__ __forceinline__ void async_copy_b128(unsigned lds_off,
                                                const __bf16* gptr) {
  unsigned long long ga = (unsigned long long)(size_t)gptr;
  asm volatile("global_load_async_to_lds_b128 %0, %1, off"
               :: "v"(lds_off), "v"(ga) : "memory");
}
__device__ __forceinline__ void wait_async0() {
  asm volatile("s_wait_asynccnt 0x0" ::: "memory");
}

// ---------------- WMMA GEMM:  C[M,N] = A[M,K] @ W[N,K]^T ----------------
// 128x128 block tile, 8 waves in 4(M)x2(N) grid, 32x64 per wave.
// Double-buffered LDS filled with global_load_async_to_lds_b128.
// mode 0: store bf16 with row remap (b*197+n) -> (b*224+n), stride Ncols
// mode 1: store f32 + bias -> outf
#define TMM 128
#define TNN 128
#define LSTR 40   // LDS row stride in bf16 elems (80B: 16B-aligned, conflict-free)

__global__ __launch_bounds__(256) void gemm_bf16_tn(
    const __bf16* __restrict__ A, const __bf16* __restrict__ W,
    int M, int Ncols, int K,
    __bf16* __restrict__ outb, float* __restrict__ outf,
    const float* __restrict__ bias, int mode)
{
  __shared__ __bf16 As[2][TMM * LSTR];
  __shared__ __bf16 Bs[2][TNN * LSTR];
  const int tid  = threadIdx.x;
  const int lane = tid & 31;
  const int wave = tid >> 5;
  const int lo = lane & 15, hi = lane >> 4;
  const int wm = wave & 3,  wn = wave >> 2;     // 4x2 -> 128x128 block tile
  const int bM = blockIdx.x, bN = blockIdx.y;

  FragC acc[2][4];
#pragma unroll
  for (int i = 0; i < 2; ++i)
#pragma unroll
    for (int j = 0; j < 4; ++j)
#pragma unroll
      for (int e = 0; e < 8; ++e) acc[i][j].f[e] = 0.f;

  // staging geometry: 4 threads per 32-col row, 64 rows per pass, 2 passes
  const int srow = tid >> 2;          // 0..63
  const int scol = (tid & 3) * 8;     // {0,8,16,24} bf16 elems

  // clamped A row (garbage rows only pollute rows we never store)
  int ra0 = bM * TMM + srow;        if (ra0 > M - 1) ra0 = M - 1;
  int ra1 = bM * TMM + srow + 64;   if (ra1 > M - 1) ra1 = M - 1;
  const int rb0 = bN * TNN + srow;        // Ncols is a multiple of 128: exact
  const int rb1 = bN * TNN + srow + 64;

  const unsigned ldsA0 = (unsigned)(size_t)&As[0][srow * LSTR + scol];
  const unsigned ldsA1 = (unsigned)(size_t)&As[0][(srow + 64) * LSTR + scol];
  const unsigned ldsB0 = (unsigned)(size_t)&Bs[0][srow * LSTR + scol];
  const unsigned ldsB1 = (unsigned)(size_t)&Bs[0][(srow + 64) * LSTR + scol];
  const unsigned bufA  = (unsigned)(TMM * LSTR * 2);   // bytes between A buffers
  const unsigned bufB  = (unsigned)(TNN * LSTR * 2);

  const int nk = K >> 5;
  // prologue: fill buffer 0 for k0 = 0
  async_copy_b128(ldsA0, A + (size_t)ra0 * K + scol);
  async_copy_b128(ldsA1, A + (size_t)ra1 * K + scol);
  async_copy_b128(ldsB0, W + (size_t)rb0 * K + scol);
  async_copy_b128(ldsB1, W + (size_t)rb1 * K + scol);
  wait_async0();
  __syncthreads();

  for (int kidx = 0; kidx < nk; ++kidx) {
    const int buf = kidx & 1;
    // kick off async fill of the other buffer for the next K slice
    if (kidx + 1 < nk) {
      const int k1 = (kidx + 1) << 5;
      const unsigned bo = (unsigned)(buf ^ 1);
      async_copy_b128(ldsA0 + bo * bufA, A + (size_t)ra0 * K + k1 + scol);
      async_copy_b128(ldsA1 + bo * bufA, A + (size_t)ra1 * K + k1 + scol);
      async_copy_b128(ldsB0 + bo * bufB, W + (size_t)rb0 * K + k1 + scol);
      async_copy_b128(ldsB1 + bo * bufB, W + (size_t)rb1 * K + k1 + scol);
    }

    FragAB a[2], b[4];
#pragma unroll
    for (int i = 0; i < 2; ++i) {
      int m = wm * 32 + i * 16 + lo;
      a[i].q[0] = *(const uint4*)(&As[buf][m * LSTR + hi * 8]);
      a[i].q[1] = *(const uint4*)(&As[buf][m * LSTR + 16 + hi * 8]);
    }
#pragma unroll
    for (int j = 0; j < 4; ++j) {
      int n = wn * 64 + j * 16 + lo;
      b[j].q[0] = *(const uint4*)(&Bs[buf][n * LSTR + hi * 8]);
      b[j].q[1] = *(const uint4*)(&Bs[buf][n * LSTR + 16 + hi * 8]);
    }
#pragma unroll
    for (int i = 0; i < 2; ++i)
#pragma unroll
      for (int j = 0; j < 4; ++j)
        acc[i][j].v = __builtin_amdgcn_wmma_f32_16x16x32_bf16(
            false, a[i].v, false, b[j].v, (short)0, acc[i][j].v, false, false);

    wait_async0();        // next buffer fully landed in LDS (this wave's share)
    __syncthreads();      // all waves' shares landed; buf free for reuse
  }

  // epilogue
#pragma unroll
  for (int i = 0; i < 2; ++i)
#pragma unroll
    for (int j = 0; j < 4; ++j) {
      int gn = bN * TNN + wn * 64 + j * 16 + lo;
#pragma unroll
      for (int e = 0; e < 8; ++e) {
        int gm = bM * TMM + wm * 32 + i * 16 + e + 8 * hi;
        if (gm < M) {
          float val = acc[i][j].f[e];
          if (mode == 0) {
            int brow = gm / SEQ;
            int nrow = gm - brow * SEQ;
            outb[((size_t)(brow * NP + nrow)) * Ncols + gn] = (__bf16)val;
          } else {
            outf[(size_t)gm * Ncols + gn] = val + bias[gn];
          }
        }
      }
    }
}

// ---------------- Fused flash attention per (b,h) ----------------
#define VSTR 232   // V^T LDS row stride (bf16): 464B rows, 16B aligned
#define PSTR 40    // P staging row stride

__global__ __launch_bounds__(256) void attn_fused(
    const __bf16* __restrict__ qkvb,          // [B][NP][3*CDIM] bf16
    const float* __restrict__ rel_table,      // [732][12]
    const int*   __restrict__ rel_index,      // [197][197]
    __bf16* __restrict__ ctxb)                // [MROWS][CDIM] bf16
{
  __shared__ __bf16 vT[HD * VSTR];            // V^T: [d][m], zero-padded m>=197
  __shared__ __bf16 pst[8 * 16 * PSTR];       // per-wave P transpose staging
  const int b = blockIdx.x / NUM_HEADS;
  const int h = blockIdx.x % NUM_HEADS;
  const int tid = threadIdx.x, lane = tid & 31, wave = tid >> 5;
  const int lo = lane & 15, hi = lane >> 4;
  const __bf16* base = qkvb + (size_t)b * NP * (3 * CDIM);

  // stage V^T (coalesced-ish read, transposed LDS write; pad with zeros)
  for (int idx = tid; idx < NP * HD; idx += 256) {
    int m = idx >> 6, d = idx & 63;
    float v = (m < SEQ) ? (float)base[(size_t)m * (3 * CDIM) + 2 * CDIM + h * HD + d]
                        : 0.f;
    vT[d * VSTR + m] = (__bf16)v;
  }
  __syncthreads();

  const float scale = 0.125f;  // 1/sqrt(64)

  for (int qt = wave; qt < 14; qt += 8) {
    // Q A-fragments (two 32-deep d chunks) straight from global
    FragAB aQ[2];
    const __bf16* qr = base + (size_t)(qt * 16 + lo) * (3 * CDIM) + h * HD;
#pragma unroll
    for (int c = 0; c < 2; ++c) {
      aQ[c].q[0] = *(const uint4*)(qr + c * 32 + hi * 8);
      aQ[c].q[1] = *(const uint4*)(qr + c * 32 + 16 + hi * 8);
    }

    float rm[8], rl[8];
#pragma unroll
    for (int e = 0; e < 8; ++e) { rm[e] = -1e30f; rl[e] = 0.f; }
    FragC o[4];
#pragma unroll
    for (int dt = 0; dt < 4; ++dt)
#pragma unroll
      for (int e = 0; e < 8; ++e) o[dt].f[e] = 0.f;

    __bf16* myp = pst + wave * 16 * PSTR;

    for (int mc = 0; mc < 7; ++mc) {           // 7 chunks of 32 key columns
      FragC s[2];
#pragma unroll
      for (int t = 0; t < 2; ++t) {
#pragma unroll
        for (int e = 0; e < 8; ++e) s[t].f[e] = 0.f;
        int mt = mc * 2 + t;
        const __bf16* kr = base + (size_t)(mt * 16 + lo) * (3 * CDIM) + CDIM + h * HD;
#pragma unroll
        for (int c = 0; c < 2; ++c) {
          FragAB bK;                            // B frag of K^T: table-order d
          bK.q[0] = *(const uint4*)(kr + c * 32 + hi * 8);
          bK.q[1] = *(const uint4*)(kr + c * 32 + 16 + hi * 8);
          s[t].v = __builtin_amdgcn_wmma_f32_16x16x32_bf16(
              false, aQ[c].v, false, bK.v, (short)0, s[t].v, false, false);
        }
        // scale + relative position bias + padding mask
        int kcol = mt * 16 + lo;
#pragma unroll
        for (int e = 0; e < 8; ++e) {
          int qrow = qt * 16 + e + 8 * hi;
          float val;
          if (kcol < SEQ && qrow < SEQ) {
            int ri = rel_index[qrow * SEQ + kcol];
            val = s[t].f[e] * scale + rel_table[ri * NUM_HEADS + h];
          } else {
            val = -1e30f;
          }
          s[t].f[e] = val;
        }
      }
      // online softmax (rows live in 16-lane halves: xor 1,2,4,8)
      float corr[8];
#pragma unroll
      for (int e = 0; e < 8; ++e) {
        float cm = fmaxf(s[0].f[e], s[1].f[e]);
#pragma unroll
        for (int off = 1; off < 16; off <<= 1)
          cm = fmaxf(cm, __shfl_xor(cm, off, 32));
        float nm = fmaxf(rm[e], cm);
        corr[e] = __expf(rm[e] - nm);
        float p0 = __expf(s[0].f[e] - nm);
        float p1 = __expf(s[1].f[e] - nm);
        s[0].f[e] = p0; s[1].f[e] = p1;
        float ps = p0 + p1;
#pragma unroll
        for (int off = 1; off < 16; off <<= 1)
          ps += __shfl_xor(ps, off, 32);
        rl[e] = rl[e] * corr[e] + ps;
        rm[e] = nm;
      }
#pragma unroll
      for (int dt = 0; dt < 4; ++dt)
#pragma unroll
        for (int e = 0; e < 8; ++e) o[dt].f[e] *= corr[e];

      // C-layout -> A-layout transpose of P through wave-private LDS
#pragma unroll
      for (int t = 0; t < 2; ++t)
#pragma unroll
        for (int e = 0; e < 8; ++e)
          myp[(e + 8 * hi) * PSTR + t * 16 + lo] = (__bf16)s[t].f[e];
      asm volatile("s_wait_dscnt 0x0" ::: "memory");   // same-wave LDS RAW
      FragAB aP;
      aP.q[0] = *(const uint4*)(&myp[lo * PSTR + hi * 8]);
      aP.q[1] = *(const uint4*)(&myp[lo * PSTR + 16 + hi * 8]);
#pragma unroll
      for (int dt = 0; dt < 4; ++dt) {
        FragAB bV;                              // B frag of V: table-order m in vT
        bV.q[0] = *(const uint4*)(&vT[(dt * 16 + lo) * VSTR + mc * 32 + hi * 8]);
        bV.q[1] = *(const uint4*)(&vT[(dt * 16 + lo) * VSTR + mc * 32 + 16 + hi * 8]);
        o[dt].v = __builtin_amdgcn_wmma_f32_16x16x32_bf16(
            false, aP.v, false, bV.v, (short)0, o[dt].v, false, false);
      }
    }
    // finalize: 1/l and store ctx (row b*197+qrow, col h*64+d)
#pragma unroll
    for (int e = 0; e < 8; ++e) rl[e] = 1.f / rl[e];
#pragma unroll
    for (int dt = 0; dt < 4; ++dt)
#pragma unroll
      for (int e = 0; e < 8; ++e) {
        int qrow = qt * 16 + e + 8 * hi;
        if (qrow < SEQ)
          ctxb[((size_t)(b * SEQ + qrow)) * CDIM + h * HD + dt * 16 + lo] =
              (__bf16)(o[dt].f[e] * rl[e]);
      }
  }
}

// ---------------- host launcher ----------------
extern "C" void kernel_launch(void* const* d_in, const int* in_sizes, int n_in,
                              void* d_out, int out_size, void* d_ws, size_t ws_size,
                              hipStream_t stream)
{
  (void)in_sizes; (void)n_in; (void)out_size; (void)ws_size;
  const float* x      = (const float*)d_in[0];
  const float* qkv_w  = (const float*)d_in[1];
  const float* proj_w = (const float*)d_in[2];
  const float* proj_b = (const float*)d_in[3];
  const float* rel_t  = (const float*)d_in[4];
  const int*   rel_i  = (const int*)d_in[5];

  char* ws = (char*)d_ws;
  size_t off = 0;
  auto alloc = [&](size_t bytes) {
    char* p = ws + off;
    off += (bytes + 255) & ~(size_t)255;
    return p;
  };

  __bf16* xb    = (__bf16*)alloc((size_t)MROWS * CDIM * 2);
  __bf16* wqkv  = (__bf16*)alloc((size_t)3 * CDIM * CDIM * 2);
  __bf16* wproj = (__bf16*)alloc((size_t)CDIM * CDIM * 2);
  __bf16* qkvb  = (__bf16*)alloc((size_t)BATCH * NP * 3 * CDIM * 2);
  __bf16* ctxb  = (__bf16*)alloc((size_t)MROWS * CDIM * 2);

  int n1 = MROWS * CDIM / 4;
  cvt_f32_bf16<<<(n1 + 255) / 256, 256, 0, stream>>>(x, xb, n1);
  int n2 = 3 * CDIM * CDIM / 4;
  cvt_f32_bf16<<<(n2 + 255) / 256, 256, 0, stream>>>(qkv_w, wqkv, n2);
  int n3 = CDIM * CDIM / 4;
  cvt_f32_bf16<<<(n3 + 255) / 256, 256, 0, stream>>>(proj_w, wproj, n3);

  dim3 g1((MROWS + TMM - 1) / TMM, (3 * CDIM) / TNN);   // 99 x 18
  gemm_bf16_tn<<<g1, 256, 0, stream>>>(xb, wqkv, MROWS, 3 * CDIM, CDIM,
                                       qkvb, nullptr, nullptr, 0);

  attn_fused<<<BATCH * NUM_HEADS, 256, 0, stream>>>(qkvb, rel_t, rel_i, ctxb);

  dim3 g2((MROWS + TMM - 1) / TMM, CDIM / TNN);         // 99 x 6
  gemm_bf16_tn<<<g2, 256, 0, stream>>>(ctxb, wproj, MROWS, CDIM, CDIM,
                                       nullptr, (float*)d_out, proj_b, 1);
}
